// KimiMLPMoE_3246995276382
// MI455X (gfx1250) — matrix operand order repaired and checked by reading, not compiled, not run on hardware
//
#include <hip/hip_runtime.h>
#include <stdint.h>

#define NUM_EXPERTS 16
#define HIDDEN 2048
#define INTER 1408
#define TOKENS 32768

typedef __attribute__((ext_vector_type(16))) __bf16 v16bf;
typedef __attribute__((ext_vector_type(2))) __bf16 v2bf;
typedef __attribute__((ext_vector_type(8))) float v8f;

union Frag {
  v16bf v;
  uint4 q[2];
};

// f32x2 -> packed bf16x2 (low half = first arg), round-to-nearest-even.
// Toolchain doesn't declare the clang builtin for gfx1250, so use the ISA
// instruction directly via inline asm (single VALU op instead of ~8).
__device__ __forceinline__ unsigned pack2(float lo, float hi) {
#if __has_builtin(__builtin_amdgcn_cvt_pk_bf16_f32)
  v2bf p = __builtin_amdgcn_cvt_pk_bf16_f32(lo, hi);
  return __builtin_bit_cast(unsigned, p);
#else
  unsigned r;
  asm("v_cvt_pk_bf16_f32 %0, %1, %2" : "=v"(r) : "v"(lo), "v"(hi));
  return r;
#endif
}

// ---------------------------------------------------------------------------
// Kernel 1: fused gate/up grouped GEMM + SwiGLU, writes h as bf16 to d_ws.
// Block tile: 128 tokens x 64 inter cols x K-step 32. 256 threads = 8 waves.
// Double-buffered LDS + register prefetch 2 tiles ahead; 1 barrier / K-step.
// ---------------------------------------------------------------------------
__global__ __launch_bounds__(256, 2) void moe_gateup_kernel(
    const float* __restrict__ X, const float* __restrict__ Wg,
    const float* __restrict__ Wu, const int* __restrict__ gs,
    unsigned short* __restrict__ Hbuf) {
  constexpr int PA = 40;  // LDS row pitch in halfs (80B => conflict-free b128)
  __shared__ unsigned short sA[2][128 * PA];
  __shared__ unsigned short sBg[2][64 * PA];
  __shared__ unsigned short sBu[2][64 * PA];

  const int nb = blockIdx.x * 64;
  const int tok0 = blockIdx.y * 128;

  // find expert owning this 128-token block (groups are block-aligned here)
  int e = 0;
  int cum = gs[0];
  while (e < NUM_EXPERTS - 1 && tok0 >= cum) { ++e; cum += gs[e]; }

  const float* Bg = Wg + (size_t)e * HIDDEN * INTER;
  const float* Bu = Wu + (size_t)e * HIDDEN * INTER;

  const int t = threadIdx.x;
  const int lane = t & 31;
  const int w = t >> 5;
  const int lhi = (lane >> 4) & 1;
  const int l16 = lane & 15;

  v8f accG[4] = {};
  v8f accU[4] = {};

  const int n0 = (t & 15) * 4;  // B-tile: 4 consecutive N per thread
  const int k0 = (t >> 4) * 2;  // B-tile: 2 consecutive K per thread

  // register staging for in-flight tile (hides global latency behind WMMA)
  float4 rA[4];
  float4 rG0, rG1, rU0, rU1;

  auto load_tiles = [&](int kk) {
#pragma unroll
    for (int i = 0; i < 4; ++i) {
      int f = t + i * 256;
      int r = f >> 3, c4 = (f & 7) * 4;
      rA[i] = *(const float4*)(X + (size_t)(tok0 + r) * HIDDEN + kk + c4);
    }
    rG0 = *(const float4*)(Bg + (size_t)(kk + k0) * INTER + nb + n0);
    rG1 = *(const float4*)(Bg + (size_t)(kk + k0 + 1) * INTER + nb + n0);
    rU0 = *(const float4*)(Bu + (size_t)(kk + k0) * INTER + nb + n0);
    rU1 = *(const float4*)(Bu + (size_t)(kk + k0 + 1) * INTER + nb + n0);
    // prefetch the weight stream further ahead (L2-resident stream)
    int kp = kk + 64;
    if (kp < HIDDEN) {
      __builtin_prefetch(Bg + (size_t)(kp + k0) * INTER + nb + n0, 0, 1);
      __builtin_prefetch(Bu + (size_t)(kp + k0) * INTER + nb + n0, 0, 1);
    }
  };

  auto store_tiles = [&](int p) {
#pragma unroll
    for (int i = 0; i < 4; ++i) {
      int f = t + i * 256;
      int r = f >> 3, c4 = (f & 7) * 4;
      uint2 pk;
      pk.x = pack2(rA[i].x, rA[i].y);
      pk.y = pack2(rA[i].z, rA[i].w);
      *(uint2*)(&sA[p][r * PA + c4]) = pk;
    }
    *(unsigned*)(&sBg[p][(n0 + 0) * PA + k0]) = pack2(rG0.x, rG1.x);
    *(unsigned*)(&sBg[p][(n0 + 1) * PA + k0]) = pack2(rG0.y, rG1.y);
    *(unsigned*)(&sBg[p][(n0 + 2) * PA + k0]) = pack2(rG0.z, rG1.z);
    *(unsigned*)(&sBg[p][(n0 + 3) * PA + k0]) = pack2(rG0.w, rG1.w);
    *(unsigned*)(&sBu[p][(n0 + 0) * PA + k0]) = pack2(rU0.x, rU1.x);
    *(unsigned*)(&sBu[p][(n0 + 1) * PA + k0]) = pack2(rU0.y, rU1.y);
    *(unsigned*)(&sBu[p][(n0 + 2) * PA + k0]) = pack2(rU0.z, rU1.z);
    *(unsigned*)(&sBu[p][(n0 + 3) * PA + k0]) = pack2(rU0.w, rU1.w);
  };

  auto compute = [&](int p) {
    Frag a;
    const unsigned short* pa = &sA[p][(w * 16 + l16) * PA + lhi * 8];
    a.q[0] = *(const uint4*)(pa);       // K = base .. base+7
    a.q[1] = *(const uint4*)(pa + 16);  // K = base+16 .. base+23
#pragma unroll
    for (int ct = 0; ct < 4; ++ct) {
      const unsigned short* pbg = &sBg[p][(ct * 16 + l16) * PA + lhi * 8];
      const unsigned short* pbu = &sBu[p][(ct * 16 + l16) * PA + lhi * 8];
      Frag bg, bu;
      bg.q[0] = *(const uint4*)(pbg);
      bg.q[1] = *(const uint4*)(pbg + 16);
      bu.q[0] = *(const uint4*)(pbu);
      bu.q[1] = *(const uint4*)(pbu + 16);
      accG[ct] = __builtin_amdgcn_wmma_f32_16x16x32_bf16(
          false, a.v, false, bg.v, (short)0, accG[ct], false, false);
      accU[ct] = __builtin_amdgcn_wmma_f32_16x16x32_bf16(
          false, a.v, false, bu.v, (short)0, accU[ct], false, false);
    }
  };

  constexpr int NIT = HIDDEN / 32;  // 64
  // pipeline prologue: tile0 -> LDS[0], tile1 -> regs
  load_tiles(0);
  store_tiles(0);
  load_tiles(32);
  __syncthreads();

  for (int i = 0; i < NIT; ++i) {
    const int p = i & 1;
    if (i + 1 < NIT) {
      store_tiles(p ^ 1);                         // tile i+1 -> other buffer
      if (i + 2 < NIT) load_tiles((i + 2) * 32);  // issue tile i+2 loads
    }
    compute(p);  // tile i
    __syncthreads();
  }

  // ---- epilogue: h = silu(gate) * up -> bf16 (fast rcp, no IEEE div) ----
  const int mrow = tok0 + w * 16 + lhi * 8;
#pragma unroll
  for (int ct = 0; ct < 4; ++ct) {
    const int ncol = nb + ct * 16 + l16;
#pragma unroll
    for (int j = 0; j < 8; ++j) {
      float g = accG[ct][j];
      float u = accU[ct][j];
      float sig = __builtin_amdgcn_rcpf(1.0f + __expf(-g));
      float h = g * sig * u;
      Hbuf[(size_t)(mrow + j) * INTER + ncol] = (unsigned short)pack2(h, h);
    }
  }
}

// ---------------------------------------------------------------------------
// Kernel 2: down grouped GEMM: out[T,H] = h[T,I](bf16) x w_down[e][I,H](f32)
// ---------------------------------------------------------------------------
__global__ __launch_bounds__(256, 2) void moe_down_kernel(
    const unsigned short* __restrict__ Hbuf, const float* __restrict__ Wd,
    const int* __restrict__ gs, float* __restrict__ Out) {
  constexpr int PA = 40;
  __shared__ unsigned short sA[2][128 * PA];
  __shared__ unsigned short sB[2][64 * PA];

  const int nb = blockIdx.x * 64;
  const int tok0 = blockIdx.y * 128;

  int e = 0;
  int cum = gs[0];
  while (e < NUM_EXPERTS - 1 && tok0 >= cum) { ++e; cum += gs[e]; }
  const float* Bw = Wd + (size_t)e * INTER * HIDDEN;

  const int t = threadIdx.x;
  const int lane = t & 31;
  const int w = t >> 5;
  const int lhi = (lane >> 4) & 1;
  const int l16 = lane & 15;

  v8f acc[4] = {};
  const int n0 = (t & 15) * 4;
  const int k0 = (t >> 4) * 2;

  uint4 rA[2];
  float4 rB0, rB1;

  auto load_tiles = [&](int kk) {
#pragma unroll
    for (int i = 0; i < 2; ++i) {
      int f = t + i * 256;
      int r = f >> 2, c8 = (f & 3) * 8;
      rA[i] = *(const uint4*)(Hbuf + (size_t)(tok0 + r) * INTER + kk + c8);
    }
    rB0 = *(const float4*)(Bw + (size_t)(kk + k0) * HIDDEN + nb + n0);
    rB1 = *(const float4*)(Bw + (size_t)(kk + k0 + 1) * HIDDEN + nb + n0);
    int kp = kk + 64;
    if (kp < INTER) {
      __builtin_prefetch(Bw + (size_t)(kp + k0) * HIDDEN + nb + n0, 0, 1);
    }
  };

  auto store_tiles = [&](int p) {
#pragma unroll
    for (int i = 0; i < 2; ++i) {
      int f = t + i * 256;
      int r = f >> 2, c8 = (f & 3) * 8;
      *(uint4*)(&sA[p][r * PA + c8]) = rA[i];
    }
    *(unsigned*)(&sB[p][(n0 + 0) * PA + k0]) = pack2(rB0.x, rB1.x);
    *(unsigned*)(&sB[p][(n0 + 1) * PA + k0]) = pack2(rB0.y, rB1.y);
    *(unsigned*)(&sB[p][(n0 + 2) * PA + k0]) = pack2(rB0.z, rB1.z);
    *(unsigned*)(&sB[p][(n0 + 3) * PA + k0]) = pack2(rB0.w, rB1.w);
  };

  auto compute = [&](int p) {
    Frag a;
    const unsigned short* pa = &sA[p][(w * 16 + l16) * PA + lhi * 8];
    a.q[0] = *(const uint4*)(pa);
    a.q[1] = *(const uint4*)(pa + 16);
#pragma unroll
    for (int ct = 0; ct < 4; ++ct) {
      const unsigned short* pb = &sB[p][(ct * 16 + l16) * PA + lhi * 8];
      Frag b;
      b.q[0] = *(const uint4*)(pb);
      b.q[1] = *(const uint4*)(pb + 16);
      acc[ct] = __builtin_amdgcn_wmma_f32_16x16x32_bf16(
          false, a.v, false, b.v, (short)0, acc[ct], false, false);
    }
  };

  constexpr int NIT = INTER / 32;  // 44
  load_tiles(0);
  store_tiles(0);
  load_tiles(32);
  __syncthreads();

  for (int i = 0; i < NIT; ++i) {
    const int p = i & 1;
    if (i + 1 < NIT) {
      store_tiles(p ^ 1);
      if (i + 2 < NIT) load_tiles((i + 2) * 32);
    }
    compute(p);
    __syncthreads();
  }

  const int mrow = tok0 + w * 16 + lhi * 8;
#pragma unroll
  for (int ct = 0; ct < 4; ++ct) {
    const int ncol = nb + ct * 16 + l16;
#pragma unroll
    for (int j = 0; j < 8; ++j) {
      Out[(size_t)(mrow + j) * HIDDEN + ncol] = acc[ct][j];
    }
  }
}

extern "C" void kernel_launch(void* const* d_in, const int* in_sizes, int n_in,
                              void* d_out, int out_size, void* d_ws,
                              size_t ws_size, hipStream_t stream) {
  (void)in_sizes;
  (void)n_in;
  (void)out_size;
  (void)ws_size;
  const float* X = (const float*)d_in[0];
  const float* Wg = (const float*)d_in[1];
  const float* Wu = (const float*)d_in[2];
  const float* Wd = (const float*)d_in[3];
  const int* gs = (const int*)d_in[4];
  unsigned short* Hbuf = (unsigned short*)d_ws;  // [TOKENS, INTER] bf16
  float* Out = (float*)d_out;

  dim3 blk(256, 1, 1);
  dim3 g1(INTER / 64, TOKENS / 128, 1);   // 22 x 256
  dim3 g2(HIDDEN / 64, TOKENS / 128, 1);  // 32 x 256

  moe_gateup_kernel<<<g1, blk, 0, stream>>>(X, Wg, Wu, gs, Hbuf);
  moe_down_kernel<<<g2, blk, 0, stream>>>(Hbuf, Wd, gs, Out);
}
